// TransformerLayer_37546604102050
// MI455X (gfx1250) — matrix operand order repaired and checked by reading, not compiled, not run on hardware
//
#include <hip/hip_runtime.h>
#include <hip/hip_bf16.h>

// ---------------------------------------------------------------------------
// Types for CDNA5 WMMA (wave32): 16x16x32 bf16 -> f32
// ---------------------------------------------------------------------------
typedef __attribute__((ext_vector_type(16))) __bf16 v16bf;
typedef __attribute__((ext_vector_type(8)))  __bf16 v8bf;
typedef __attribute__((ext_vector_type(8)))  float  v8f;

// Async global->LDS (gfx1250 ASYNCcnt path), guarded so either toolchain compiles.
#if defined(__AMDGCN__) && __has_builtin(__builtin_amdgcn_global_load_async_to_lds_b128) && __has_builtin(__builtin_amdgcn_s_wait_asynccnt)
#define USE_ASYNC_LDS 1
#else
#define USE_ASYNC_LDS 0
#endif

#if USE_ASYNC_LDS
typedef __attribute__((ext_vector_type(4))) int v4i_t;
typedef __attribute__((address_space(1))) v4i_t* as1_v4i;   // global int4*
typedef __attribute__((address_space(3))) v4i_t* as3_v4i;   // LDS int4*
#endif

// Copy 16 bytes (8 bf16) from global to LDS, per-lane.
__device__ inline void copy16_g2l(const __bf16* g, __bf16* l) {
#if USE_ASYNC_LDS
  __builtin_amdgcn_global_load_async_to_lds_b128((as1_v4i)g, (as3_v4i)l, 0, 0);
#else
  *(v8bf*)l = *(const v8bf*)g;
#endif
}
__device__ inline void g2l_wait() {
#if USE_ASYNC_LDS
  __builtin_amdgcn_s_wait_asynccnt(0);
#endif
}

__device__ inline v8f wmma_bf16(v16bf a, v16bf b, v8f c) {
  // 8 args: (neg_a, A, neg_b, B, c_mod, C, reuse_a, reuse_b)
  return __builtin_amdgcn_wmma_f32_16x16x32_bf16(false, a, false, b,
                                                 (short)0, c, false, false);
}

// A-fragment: 16x32 bf16, row m = lane&15; lane<16 holds K 0..7 & 16..23,
// lane>=16 holds K 8..15 & 24..31 (ISA 7.12.2). src is row-major [row][k].
__device__ inline v16bf load_afrag(const __bf16* src, int stride, int row0, int k0, int lane) {
  int m  = lane & 15;
  int kb = (lane < 16) ? 0 : 8;
  const __bf16* p = src + (size_t)(row0 + m) * stride + k0 + kb;
  v8bf lo = *(const v8bf*)(p);
  v8bf hi = *(const v8bf*)(p + 16);
  return __builtin_shufflevector(lo, hi, 0,1,2,3,4,5,6,7,8,9,10,11,12,13,14,15);
}

// B-fragment: 32x16 bf16, col n = lane&15; lane<16 holds K 0..7 & 16..23, lane>=16 K 8..15 & 24..31.
// src is K-contiguous per column: [col][k] (transposed tile).
__device__ inline v16bf load_bfrag(const __bf16* src, int stride, int col0, int k0, int lane) {
  int n  = lane & 15;
  int kb = (lane < 16) ? 0 : 8;
  const __bf16* p = src + (size_t)(col0 + n) * stride + k0 + kb;
  v8bf lo = *(const v8bf*)(p);
  v8bf hi = *(const v8bf*)(p + 16);
  return __builtin_shufflevector(lo, hi, 0,1,2,3,4,5,6,7,8,9,10,11,12,13,14,15);
}

// ---------------------------------------------------------------------------
// Weight conversion with transpose: in fp32 [K][N] -> out bf16 [N][K]
// ---------------------------------------------------------------------------
__global__ __launch_bounds__(256) void cvt_t_bf16_kernel(const float* __restrict__ in,
                                                         __bf16* __restrict__ out,
                                                         int K, int N) {
  int idx = blockIdx.x * 256 + threadIdx.x;
  if (idx < K * N) {
    int k = idx / N, n = idx - k * N;
    out[(size_t)n * K + k] = (__bf16)in[idx];
  }
}

// ---------------------------------------------------------------------------
// LayerNorm over D, fused bf16 cast. One 256-thread block per row.
// ---------------------------------------------------------------------------
__global__ __launch_bounds__(256) void ln_bf16_kernel(const float* __restrict__ x,
                                                      const float* __restrict__ g,
                                                      const float* __restrict__ b,
                                                      __bf16* __restrict__ out, int D) {
  __shared__ float red[256];
  const int tid = threadIdx.x;
  const float* xr = x + (size_t)blockIdx.x * D;

  float s = 0.f;
  for (int i = tid; i < D; i += 256) s += xr[i];
  red[tid] = s; __syncthreads();
  for (int off = 128; off > 0; off >>= 1) {
    if (tid < off) red[tid] += red[tid + off];
    __syncthreads();
  }
  const float mean = red[0] / (float)D;
  __syncthreads();

  float v = 0.f;
  for (int i = tid; i < D; i += 256) { float d = xr[i] - mean; v += d * d; }
  red[tid] = v; __syncthreads();
  for (int off = 128; off > 0; off >>= 1) {
    if (tid < off) red[tid] += red[tid + off];
    __syncthreads();
  }
  const float rstd = rsqrtf(red[0] / (float)D + 1e-5f);

  __bf16* orow = out + (size_t)blockIdx.x * D;
  for (int i = tid; i < D; i += 256)
    orow[i] = (__bf16)((xr[i] - mean) * rstd * g[i] + b[i]);
}

// ---------------------------------------------------------------------------
// bf16 WMMA GEMM: C[M,N] = A[M,K] @ Bt[N,K]^T + bias (+ReLU)
// MODE 0: fp32 out + fp32 residual; MODE 1: bf16 row-major out;
// MODE 2: bf16 out scattered to V-transposed layout [b][h][dh][s].
// Block tile 128x256, K-step 32, double-buffered LDS, async global->LDS.
// 8 waves (2x4), each wave 64x64 = 16 WMMA tiles (1.0 LDS frag reads / WMMA).
// ---------------------------------------------------------------------------
template<int MODE, bool RELU>
__global__ __launch_bounds__(256) void gemm_bf16_kernel(
    const __bf16* __restrict__ A, const __bf16* __restrict__ Bt,
    const float* __restrict__ bias, const float* __restrict__ resid,
    float* __restrict__ outF, __bf16* __restrict__ outB,
    int M, int N, int K, int Sdim, int Bdim)
{
  constexpr int BM = 128, BN = 256, BK = 32;
  __shared__ __align__(64) __bf16 As[2][BM * BK];   // [row][k]  stride BK, 16 KB
  __shared__ __align__(64) __bf16 Bs[2][BN * BK];   // [col][k]  stride BK, 32 KB

  const int tid  = threadIdx.x;
  const int lane = tid & 31;
  const int wid  = tid >> 5;
  const int wm   = wid >> 2;          // 0..1  -> 64-row slab
  const int wn   = wid & 3;           // 0..3  -> 64-col slab
  const int rowBase = blockIdx.y * BM;
  const int colBase = blockIdx.x * BN;

  v8f acc[4][4];
#pragma unroll
  for (int mt = 0; mt < 4; ++mt)
#pragma unroll
    for (int nt = 0; nt < 4; ++nt) acc[mt][nt] = (v8f){};

  const int lr = tid >> 2;            // 0..63  (row of A tile / col of B tile)
  const int lc = (tid & 3) * 8;       // 0..24  (k offset)

  auto issue_tile = [&](int buf, int k0) {
#pragma unroll
    for (int i = 0; i < 2; ++i) {     // A: 128 rows
      const int r = lr + i * 64;
      copy16_g2l(A + (size_t)(rowBase + r) * K + k0 + lc, &As[buf][r * BK + lc]);
    }
#pragma unroll
    for (int i = 0; i < 4; ++i) {     // B: 256 cols
      const int r = lr + i * 64;
      copy16_g2l(Bt + (size_t)(colBase + r) * K + k0 + lc, &Bs[buf][r * BK + lc]);
    }
  };

  issue_tile(0, 0);

  for (int k0 = 0; k0 < K; k0 += BK) {
    const int cur = (k0 / BK) & 1;
    g2l_wait();              // current tile landed in LDS (async path)
    __syncthreads();         // all waves done with buffer `cur^1`, tile `cur` visible
    if (k0 + BK < K) issue_tile(cur ^ 1, k0 + BK);   // prefetch next tile

    v16bf af[4], bf[4];
#pragma unroll
    for (int mt = 0; mt < 4; ++mt) af[mt] = load_afrag(As[cur], BK, wm * 64 + mt * 16, 0, lane);
#pragma unroll
    for (int nt = 0; nt < 4; ++nt) bf[nt] = load_bfrag(Bs[cur], BK, wn * 64 + nt * 16, 0, lane);
#pragma unroll
    for (int mt = 0; mt < 4; ++mt)
#pragma unroll
      for (int nt = 0; nt < 4; ++nt)
        acc[mt][nt] = wmma_bf16(af[mt], bf[nt], acc[mt][nt]);
    __syncthreads();
  }

  // Epilogue. C layout: VGPR j holds row (j + 8*(lane>=16)), col = lane&15.
  const int nl = lane & 15;
  const int mh = (lane < 16) ? 0 : 8;
#pragma unroll
  for (int mt = 0; mt < 4; ++mt) {
#pragma unroll
    for (int nt = 0; nt < 4; ++nt) {
      const int col = colBase + wn * 64 + nt * 16 + nl;
      const float bv = bias[col];
#pragma unroll
      for (int j = 0; j < 8; ++j) {
        const int row = rowBase + wm * 64 + mt * 16 + mh + j;
        float val = acc[mt][nt][j] + bv;
        if (RELU) val = fmaxf(val, 0.f);
        if (MODE == 0) {
          val += resid[(size_t)row * N + col];
          outF[(size_t)row * N + col] = val;
        } else if (MODE == 1) {
          outB[(size_t)row * N + col] = (__bf16)val;
        } else { // MODE 2: row = s*B + b  ->  vt[(b*N + col)*S + s]
          const int b_ = row % Bdim;
          const int s_ = row / Bdim;
          outB[((size_t)b_ * N + col) * Sdim + s_] = (__bf16)val;
        }
      }
    }
  }
}

// ---------------------------------------------------------------------------
// Flash-style attention. Grid: (S/128, B*H). 8 waves, each owns 16 query rows.
// q,k: [s, b, h*64+dh] bf16 ; vt: [b, h, dh, s] bf16 (pre-transposed by GEMM).
// ---------------------------------------------------------------------------
__global__ __launch_bounds__(256) void attn_kernel(
    const __bf16* __restrict__ q, const __bf16* __restrict__ k,
    const __bf16* __restrict__ vt, __bf16* __restrict__ ctx,
    int S, int B, int H, int DH)
{
  constexpr int BQ = 128, BKV = 64, D_H = 64;
  __shared__ __align__(64) __bf16 Qs[BQ * D_H];    // [q][dh]   16 KB
  __shared__ __align__(64) __bf16 Ks[BKV * D_H];   // [key][dh]  8 KB
  __shared__ __align__(64) __bf16 Vs[D_H * BKV];   // [dh][key]  8 KB
  __shared__ __align__(64) __bf16 Ps[BQ * BKV];    // [q][key]  16 KB

  const int tid  = threadIdx.x;
  const int lane = tid & 31;
  const int w    = tid >> 5;                 // wave id: rows [w*16, w*16+16)
  const int bh   = blockIdx.y;
  const int b    = bh / H;
  const int h    = bh - b * H;
  const int q0   = blockIdx.x * BQ;
  const size_t rowStride = (size_t)B * H * DH;          // = B*D
  const size_t base      = (size_t)b * H * DH + (size_t)h * DH;
  const size_t baseV     = ((size_t)b * H + h) * DH * S;

  // Load Q block (128 x 64)
  {
    const int r = tid >> 3;                 // 0..31
    const int c = (tid & 7) * 8;            // 0..56
#pragma unroll
    for (int i = 0; i < 4; ++i) {
      const int rr = r + i * 32;
      copy16_g2l(q + base + (size_t)(q0 + rr) * rowStride + c, &Qs[rr * D_H + c]);
    }
  }

  float mrow[8], lrow[8];
  v8f o[4];
#pragma unroll
  for (int j = 0; j < 8; ++j) { mrow[j] = -1e30f; lrow[j] = 0.f; }
#pragma unroll
  for (int ot = 0; ot < 4; ++ot) o[ot] = (v8f){};

  const float scale = 0.125f;               // 1/sqrt(64)
  const int nl = lane & 15;
  const int mh = (lane < 16) ? 0 : 8;

  for (int kb = 0; kb < S; kb += BKV) {
    __syncthreads();                        // Ks/Vs free to overwrite
    {
      const int r = tid >> 3;               // 0..31
      const int c = (tid & 7) * 8;          // 0..56
#pragma unroll
      for (int i = 0; i < 2; ++i) {
        const int rr = r + i * 32;          // key 0..63 / dh 0..63
        copy16_g2l(k  + base  + (size_t)(kb + rr) * rowStride + c, &Ks[rr * D_H + c]);
        copy16_g2l(vt + baseV + (size_t)rr * S + kb + c,           &Vs[rr * BKV + c]);
      }
    }
    g2l_wait();
    __syncthreads();

    // S = Q K^T  (16 x 64) -> 4 tiles of 16x16, K-dim 64 = 2 WMMA steps
    v8f sc[4];
#pragma unroll
    for (int nt = 0; nt < 4; ++nt) {
      sc[nt] = (v8f){};
#pragma unroll
      for (int kk = 0; kk < D_H; kk += 32) {
        v16bf af = load_afrag(Qs, D_H, w * 16, kk, lane);
        v16bf bf = load_bfrag(Ks, D_H, nt * 16, kk, lane);
        sc[nt] = wmma_bf16(af, bf, sc[nt]);
      }
    }

    // online softmax (row m = j + mh lives in VGPR j; reduce across 16 lanes)
#pragma unroll
    for (int nt = 0; nt < 4; ++nt)
#pragma unroll
      for (int j = 0; j < 8; ++j) sc[nt][j] *= scale;

#pragma unroll
    for (int j = 0; j < 8; ++j) {
      float mx = sc[0][j];
#pragma unroll
      for (int nt = 1; nt < 4; ++nt) mx = fmaxf(mx, sc[nt][j]);
      mx = fmaxf(mx, __shfl_xor(mx, 1));
      mx = fmaxf(mx, __shfl_xor(mx, 2));
      mx = fmaxf(mx, __shfl_xor(mx, 4));
      mx = fmaxf(mx, __shfl_xor(mx, 8));
      const float mnew = fmaxf(mrow[j], mx);

      float sum = 0.f;
#pragma unroll
      for (int nt = 0; nt < 4; ++nt) {
        const float p = __expf(sc[nt][j] - mnew);
        sc[nt][j] = p;
        sum += p;
      }
      sum += __shfl_xor(sum, 1);
      sum += __shfl_xor(sum, 2);
      sum += __shfl_xor(sum, 4);
      sum += __shfl_xor(sum, 8);

      const float f = __expf(mrow[j] - mnew);
      lrow[j] = lrow[j] * f + sum;
      mrow[j] = mnew;
#pragma unroll
      for (int ot = 0; ot < 4; ++ot) o[ot][j] *= f;
    }

    // P -> LDS (bf16)
#pragma unroll
    for (int nt = 0; nt < 4; ++nt)
#pragma unroll
      for (int j = 0; j < 8; ++j)
        Ps[(w * 16 + mh + j) * BKV + nt * 16 + nl] = (__bf16)sc[nt][j];
    __syncthreads();

    // O += P V  (16 x 64), K-dim = 64 keys = 2 WMMA steps
#pragma unroll
    for (int kk = 0; kk < BKV; kk += 32) {
      v16bf pa = load_afrag(Ps, BKV, w * 16, kk, lane);
#pragma unroll
      for (int ot = 0; ot < 4; ++ot) {
        v16bf vb = load_bfrag(Vs, BKV, ot * 16, kk, lane);
        o[ot] = wmma_bf16(pa, vb, o[ot]);
      }
    }
  }

  // normalize and write ctx (bf16) at [s, b, h*64 + dh]
#pragma unroll
  for (int ot = 0; ot < 4; ++ot)
#pragma unroll
    for (int j = 0; j < 8; ++j) {
      const int srow = q0 + w * 16 + mh + j;
      const float val = o[ot][j] / lrow[j];
      ctx[base + (size_t)srow * rowStride + ot * 16 + nl] = (__bf16)val;
    }
}

// ---------------------------------------------------------------------------
// Host-side orchestration
// ---------------------------------------------------------------------------
extern "C" void kernel_launch(void* const* d_in, const int* in_sizes, int n_in,
                              void* d_out, int out_size, void* d_ws, size_t ws_size,
                              hipStream_t stream) {
  (void)in_sizes; (void)n_in; (void)out_size; (void)ws_size;
  constexpr int S = 2048, B = 4, D = 1024, H = 16, DH = 64, F = 4096;
  constexpr int M = S * B;

  const float* x     = (const float*)d_in[0];
  const float* ln1_g = (const float*)d_in[1];
  const float* ln1_b = (const float*)d_in[2];
  const float* ln2_g = (const float*)d_in[3];
  const float* ln2_b = (const float*)d_in[4];
  const float* Wq = (const float*)d_in[5];  const float* bq = (const float*)d_in[6];
  const float* Wk = (const float*)d_in[7];  const float* bk = (const float*)d_in[8];
  const float* Wv = (const float*)d_in[9];  const float* bv = (const float*)d_in[10];
  const float* Wo = (const float*)d_in[11]; const float* bo = (const float*)d_in[12];
  const float* W1 = (const float*)d_in[13]; const float* b1 = (const float*)d_in[14];
  const float* W2 = (const float*)d_in[15]; const float* b2 = (const float*)d_in[16];
  float* out = (float*)d_out;

  char* ws = (char*)d_ws;
  size_t off = 0;
  auto alloc_bf = [&](size_t n) { __bf16* p = (__bf16*)(ws + off); off += n * sizeof(__bf16); return p; };
  auto alloc_f  = [&](size_t n) { float*  p = (float*)(ws + off);  off += n * sizeof(float);  return p; };

  __bf16* hb   = alloc_bf((size_t)M * D);
  __bf16* qb   = alloc_bf((size_t)M * D);
  __bf16* kbuf = alloc_bf((size_t)M * D);
  __bf16* vtb  = alloc_bf((size_t)M * D);   // [b][h][dh][s]
  __bf16* ctxb = alloc_bf((size_t)M * D);
  float*  x1   = alloc_f ((size_t)M * D);
  __bf16* h2b  = alloc_bf((size_t)M * D);
  __bf16* f1b  = alloc_bf((size_t)M * F);
  __bf16* Wqt  = alloc_bf((size_t)D * D);   // all weights stored transposed [N][K]
  __bf16* Wkt  = alloc_bf((size_t)D * D);
  __bf16* Wvt  = alloc_bf((size_t)D * D);
  __bf16* Wot  = alloc_bf((size_t)D * D);
  __bf16* W1t  = alloc_bf((size_t)F * D);
  __bf16* W2t  = alloc_bf((size_t)D * F);

  // 1) weights -> bf16, transposed to [N][K]
  cvt_t_bf16_kernel<<<(D * D) / 256, 256, 0, stream>>>(Wq, Wqt, D, D);
  cvt_t_bf16_kernel<<<(D * D) / 256, 256, 0, stream>>>(Wk, Wkt, D, D);
  cvt_t_bf16_kernel<<<(D * D) / 256, 256, 0, stream>>>(Wv, Wvt, D, D);
  cvt_t_bf16_kernel<<<(D * D) / 256, 256, 0, stream>>>(Wo, Wot, D, D);
  cvt_t_bf16_kernel<<<(D * F) / 256, 256, 0, stream>>>(W1, W1t, D, F);
  cvt_t_bf16_kernel<<<(F * D) / 256, 256, 0, stream>>>(W2, W2t, F, D);

  // 2) h = LN1(x) -> bf16
  ln_bf16_kernel<<<M, 256, 0, stream>>>(x, ln1_g, ln1_b, hb, D);

  // 3) q,k,v projections (v written in transposed [b,h,dh,s] layout)
  dim3 gQKV(D / 256, M / 128);
  gemm_bf16_kernel<1, false><<<gQKV, 256, 0, stream>>>(hb, Wqt, bq, nullptr, nullptr, qb,   M, D, D, S, B);
  gemm_bf16_kernel<1, false><<<gQKV, 256, 0, stream>>>(hb, Wkt, bk, nullptr, nullptr, kbuf, M, D, D, S, B);
  gemm_bf16_kernel<2, false><<<gQKV, 256, 0, stream>>>(hb, Wvt, bv, nullptr, nullptr, vtb,  M, D, D, S, B);

  // 4) attention -> ctx (bf16)
  attn_kernel<<<dim3(S / 128, B * H), 256, 0, stream>>>(qb, kbuf, vtb, ctxb, S, B, H, DH);

  // 5) x1 = x + ctx @ Wo + bo   (fp32)
  gemm_bf16_kernel<0, false><<<gQKV, 256, 0, stream>>>(ctxb, Wot, bo, x, x1, nullptr, M, D, D, S, B);

  // 6) h2 = LN2(x1) -> bf16
  ln_bf16_kernel<<<M, 256, 0, stream>>>(x1, ln2_g, ln2_b, h2b, D);

  // 7) f1 = relu(h2 @ W1 + b1) -> bf16
  gemm_bf16_kernel<1, true><<<dim3(F / 256, M / 128), 256, 0, stream>>>(h2b, W1t, b1, nullptr, nullptr, f1b, M, F, D, S, B);

  // 8) out = x1 + f1 @ W2 + b2  (fp32)
  gemm_bf16_kernel<0, false><<<dim3(D / 256, M / 128), 256, 0, stream>>>(f1b, W2t, b2, x1, out, nullptr, M, D, F, S, B);
}
